// CoupledTwistorLNN_5660766896351
// MI455X (gfx1250) — matrix operand order repaired and checked by reading, not compiled, not run on hardware
//
#include <hip/hip_runtime.h>
#include <hip/hip_bf16.h>

typedef __attribute__((ext_vector_type(16))) _Float16 v16h;
typedef __attribute__((ext_vector_type(8)))  _Float16 v8h;
typedef __attribute__((ext_vector_type(4)))  _Float16 v4h;
typedef __attribute__((ext_vector_type(8)))  float    v8f;
typedef __attribute__((ext_vector_type(4)))  float    v4f;

// Global (addrspace 1) views — guarantees global_load/global_store (LOADcnt/STOREcnt
// only) instead of flat ops that also tick DScnt and force joint counter drains.
#define AS1 __attribute__((address_space(1)))
typedef AS1 const _Float16 g_cf16;
typedef AS1 const v16h    g_cv16h;
typedef AS1 const v4f     g_cv4f;
typedef AS1 float         g_f32;

#define T_STEPS 512
#define BATCH   256
#define INPD    128
#define HIDD    256
#define OUTD    64
#define DTC     0.1f
#define CPL     0.1f

// f16 weight workspace layout (element offsets into _Float16* ws16).
// The 7 recurrent 256x256 matrices are stored INTERLEAVED at B-fragment
// granularity: element (mat, n, k) lives at (n*56 + (k/32)*7 + mat)*32 + (k%32),
// so one lane's 7 fragments for a K-block are 448 contiguous bytes.
#define NMAT        7
#define WOFF_REC    0                 // 7 * 256*256 interleaved
#define WOFF_UH     458752            // 256x128
#define WOFF_UZ     491520            // 256x128
#define WOFF_OUTH   524288            // 64x256
#define WOFF_OUTZ   540672            // 64x256
#define WS_HALFS    557056
#define WS_F16_BYTES (WS_HALFS * 2)   // 1,114,112 bytes; f32 combined biases follow
// mat indices within the interleaved block (load/issue order):
//   0=W_tau_h  1=W_h  2=CPL*h2z  3=W_z_real  4=CPL*z2h  5=W_z_imag  6=W_tau_z

static __device__ __forceinline__ v8f wmma_f16(v16h a, v16h b, v8f c) {
    return __builtin_amdgcn_wmma_f32_16x16x32_f16(false, a, false, b, (short)0, c, false, false);
}

static __device__ __forceinline__ v16h mk16(v8h lo, v8h hi) {
    return __builtin_shufflevector(lo, hi, 0,1,2,3,4,5,6,7,8,9,10,11,12,13,14,15);
}

// A-fragment (16x32 f16, MxK) from LDS: lane group 0 holds K = k0+[0..15],
// group 1 holds K = k0+[16..31]; row = lane&15.
static __device__ __forceinline__ v16h afrag(const _Float16* base, int row, int rowstride,
                                             int k0, int grp) {
    const int klo = k0 + grp * 16;
    const int khi = k0 + grp * 16 + 8;
    v8h lo = *(const v8h*)(base + row * rowstride + klo);
    v8h hi = *(const v8h*)(base + row * rowstride + khi);
    return mk16(lo, hi);
}

// B-fragment (32x16 f16, KxN) from global: p already includes base + lane part;
// 'off' is a compile-time element offset folded into the load's immediate.
static __device__ __forceinline__ v16h bfrag(g_cf16* p, int off) {
    return *(g_cv16h*)(p + off);
}

static __device__ __forceinline__ float frcp(float x) {
    return __builtin_amdgcn_rcpf(x);
}

static __device__ __forceinline__ float fast_tanh(float x) {
    // tanh(x) = 1 - 2/(exp(2x)+1);  inf-safe at both ends.
    float e = __builtin_amdgcn_exp2f(x * (2.0f * 1.4426950408889634f));
    return 1.0f - 2.0f * frcp(e + 1.0f);
}

static __device__ __forceinline__ float tau_fn(float x) {
    // clip(sigmoid(x), 0.01, 1.0) + 1e-6
    float e = __builtin_amdgcn_exp2f(-x * 1.4426950408889634f);
    float s = frcp(1.0f + e);
    s = fminf(fmaxf(s, 0.01f), 1.0f);
    return s + 1e-6f;
}

__global__ void ctl_prep_kernel(const float* Wh, const float* Wtauh, const float* Wzr,
                                const float* Wzi, const float* Wtauz,
                                const float* h2z, const float* z2h,
                                const float* Uh, const float* Uz,
                                const float* outh, const float* outz,
                                const float* Uh_b, const float* b_h, const float* z2h_b,
                                const float* Uz_b, const float* h2z_b,
                                _Float16* ws16, float* wsf) {
    const int i = blockIdx.x * blockDim.x + threadIdx.x;
    const int stride = gridDim.x * blockDim.x;
    for (int j = i; j < 65536; j += stride) {
        const int n = j >> 8, k = j & 255;
        const int base = WOFF_REC + (n * 56 + (k >> 5) * NMAT) * 32 + (k & 31);
        ws16[base + 0 * 32] = (_Float16)Wtauh[j];
        ws16[base + 1 * 32] = (_Float16)Wh[j];
        ws16[base + 2 * 32] = (_Float16)(CPL * h2z[j]);
        ws16[base + 3 * 32] = (_Float16)Wzr[j];
        ws16[base + 4 * 32] = (_Float16)(CPL * z2h[j]);
        ws16[base + 5 * 32] = (_Float16)Wzi[j];
        ws16[base + 6 * 32] = (_Float16)Wtauz[j];
    }
    for (int j = i; j < 32768; j += stride) {
        ws16[WOFF_UH + j] = (_Float16)Uh[j];
        ws16[WOFF_UZ + j] = (_Float16)Uz[j];
    }
    for (int j = i; j < 16384; j += stride) {
        ws16[WOFF_OUTH + j] = (_Float16)outh[j];
        ws16[WOFF_OUTZ + j] = (_Float16)outz[j];
    }
    for (int j = i; j < 256; j += stride) {
        wsf[j]       = Uh_b[j] + b_h[j] + CPL * z2h_b[j];   // h linear combined bias
        wsf[256 + j] = Uz_b[j] + CPL * h2z_b[j];            // z linear combined bias
    }
}

// Exactly one 512-thread block per WGP (grid = 16) -> 4 waves/SIMD is the hard
// ceiling; tell the backend so it spends the full register budget on keeping
// loads in flight instead of optimizing for unreachable occupancy.
__global__ __launch_bounds__(512) __attribute__((amdgpu_waves_per_eu(4, 4)))
void ctl_recurrent_kernel(const float* __restrict__ xg,
                          const _Float16* __restrict__ ws16,
                          const float* __restrict__ wsf,
                          const float* __restrict__ Wh_b,
                          const float* __restrict__ Wtauh_b,
                          const float* __restrict__ Wzr_b,
                          const float* __restrict__ Wzi_b,
                          const float* __restrict__ Wtauz_b,
                          const float* __restrict__ bzr_v,
                          const float* __restrict__ bzi_v,
                          const float* __restrict__ outh_b,
                          const float* __restrict__ outz_b,
                          float* __restrict__ yout) {
    __shared__ _Float16 h16s[16 * HIDD];
    __shared__ _Float16 zr16s[16 * HIDD];
    __shared__ _Float16 zi16s[16 * HIDD];
    __shared__ _Float16 za16s[16 * HIDD];
    __shared__ _Float16 x16s[16 * INPD];

    const int tid  = threadIdx.x;
    const int wave = tid >> 5;        // 0..15 -> N tile (16 cols each) of HID
    const int lane = tid & 31;
    const int grp  = lane >> 4;       // 0/1
    const int ln   = lane & 15;
    const int b0   = blockIdx.x * 16; // batch row base for this workgroup
    const int nc   = wave * 16 + ln;  // this lane's output column (C/D layout)

    // per-lane element offsets into weight storage (B-fragment addressing)
    const int loRec = nc * (56 * 32) + grp * 16;  // interleaved recurrent block
    const int lo128 = nc * INPD + grp * 16;       // K=128 matrices
    const int no    = (wave & 3) * 16 + ln;       // out-proj column
    const int loOut = no * HIDD + grp * 16;

    g_cv4f* xgv = (g_cv4f*)xg;        // global-AS view of x, 16B vectors
    g_f32*  yg  = (g_f32*)yout;       // global-AS view of y

    // zero-init LDS state
    for (int e = tid; e < 16 * HIDD; e += 512) {
        h16s[e] = (_Float16)0.f; zr16s[e] = (_Float16)0.f;
        zi16s[e] = (_Float16)0.f; za16s[e] = (_Float16)0.f;
    }
    // load x[0]: 512 threads, one float4 -> v4h each (16 rows x 32 chunks)
    {
        const int m = tid >> 5, c = tid & 31;
        const v4f v = xgv[(size_t)(b0 + m) * (INPD / 4) + c];
        v4h hv; hv[0] = (_Float16)v[0]; hv[1] = (_Float16)v[1];
        hv[2] = (_Float16)v[2]; hv[3] = (_Float16)v[3];
        *(v4h*)(x16s + m * INPD + c * 4) = hv;
    }

    // per-lane column-biases (constant over time)
    const float bTauH = Wtauh_b[nc];
    const float bWh   = Wh_b[nc];
    const float bHL   = wsf[nc];            // U_h_b + b_h + CPL*z2h_b
    const float bTauZ = Wtauz_b[nc];
    const float bWzr  = Wzr_b[nc];
    const float bWzi  = Wzi_b[nc];
    const float bZL   = wsf[256 + nc];      // U_z_b + CPL*h2z_b
    const float bZr   = bzr_v[nc];
    const float bZi   = bzi_v[nc];
    const float bOut  = (wave < 4) ? (outh_b[no] + outz_b[no]) : 0.f;

    // persistent f32 state in WMMA C/D layout: element r -> (m = r + 8*grp, n = nc)
    v8f hS = {}, zrS = {}, ziS = {};

    __syncthreads();

    for (int t = 0; t < T_STEPS; ++t) {
        // Opaque *integer* offset: forces weight loads to stay inside the t-loop
        // (no hoist -> no 500-VGPR spill), while the base pointer keeps its
        // global address space so loads lower to global_load_b128 (LOADcnt only,
        // decoupled from the LDS DScnt stream).
        size_t toff = 0;
        asm volatile("" : "+s"(toff));
        g_cf16* wb  = (g_cf16*)ws16 + toff;
        g_cf16* pB  = wb + WOFF_REC  + loRec;   // interleaved recurrent weights
        g_cf16* pUh = wb + WOFF_UH   + lo128;
        g_cf16* pUz = wb + WOFF_UZ   + lo128;
        g_cf16* pOh = wb + WOFF_OUTH + loOut;
        g_cf16* pOz = wb + WOFF_OUTZ + loOut;

        v8f aTH = {}, aWH = {}, aHL = {}, aTZ = {}, aZR = {}, aZI = {}, aZL = {};

        // ---- input GEMMs (K=128), separate regular loop, chained into aHL/aZL ----
#pragma unroll
        for (int kb = 0; kb < 4; ++kb) {
            const int k0 = kb * 32;
            v16h Buh = bfrag(pUh, k0);
            v16h Buz = bfrag(pUz, k0);
            v16h Ax  = afrag(x16s, ln, INPD, k0, grp);
            aHL = wmma_f16(Ax, Buh, aHL);
            aZL = wmma_f16(Ax, Buz, aZL);
        }

        // ---- recurrent GEMMs (K=256): the 7 fragments per K-block are 448
        // contiguous bytes per lane (one clause, ~4 cachelines); WMMAs issue in
        // load order so staged partial loadcnt waits can overlap ----
#pragma unroll
        for (int kb = 0; kb < 8; ++kb) {
            const int ko = kb * (NMAT * 32);
            v16h Bth = bfrag(pB, ko + 0 * 32);
            v16h Bwh = bfrag(pB, ko + 1 * 32);
            v16h Bh2 = bfrag(pB, ko + 2 * 32);
            v16h Bzr = bfrag(pB, ko + 3 * 32);
            v16h Bz2 = bfrag(pB, ko + 4 * 32);
            v16h Bzi = bfrag(pB, ko + 5 * 32);
            v16h Btz = bfrag(pB, ko + 6 * 32);
            const int k0 = kb * 32;
            v16h Ah  = afrag(h16s,  ln, HIDD, k0, grp);
            v16h Azr = afrag(zr16s, ln, HIDD, k0, grp);
            v16h Azi = afrag(zi16s, ln, HIDD, k0, grp);
            v16h Aza = afrag(za16s, ln, HIDD, k0, grp);

            aTH = wmma_f16(Ah,  Bth, aTH);
            aWH = wmma_f16(Ah,  Bwh, aWH);
            aZL = wmma_f16(Ah,  Bh2, aZL);   // + CPL*h@h2z^T
            aZR = wmma_f16(Azr, Bzr, aZR);
            aHL = wmma_f16(Azr, Bz2, aHL);   // + CPL*zr@z2h^T
            aZI = wmma_f16(Azi, Bzi, aZI);
            aTZ = wmma_f16(Aza, Btz, aTZ);
        }
        __syncthreads();  // all LDS reads of state(t)/x(t) done

        // ---------------- elementwise state update (registers) ----------------
#pragma unroll
        for (int r = 0; r < 8; ++r) {
            const float h  = hS[r];
            const float zr = zrS[r];
            const float zi = ziS[r];
            const float rth = DTC * frcp(tau_fn(aTH[r] + bTauH));
            const float hn = h + rth * (-h + fast_tanh(aWH[r] + bWh) + aHL[r] + bHL);
            const float rtz = DTC * frcp(tau_fn(aTZ[r] + bTauZ));
            const float lin = aZL[r] + bZL;
            const float zrn = zr + rtz * (-zr + fast_tanh(aZR[r] + bWzr) + lin + bZr);
            const float zin = zi + rtz * (-zi + fast_tanh(aZI[r] + bWzi) + lin + bZi);
            hS[r] = hn; zrS[r] = zrn; ziS[r] = zin;
            const float zab = sqrtf(zrn * zrn + zin * zin + 1e-24f);
            const int addr = (r + 8 * grp) * HIDD + nc;
            h16s[addr]  = (_Float16)hn;
            zr16s[addr] = (_Float16)zrn;
            zi16s[addr] = (_Float16)zin;
            za16s[addr] = (_Float16)zab;
        }
        // prefetch next x slice into LDS (one 16B global load -> 8B ds store per thread)
        if (t + 1 < T_STEPS) {
            const int m = tid >> 5, c = tid & 31;
            const v4f v = xgv[((size_t)(t + 1) * BATCH + b0 + m) * (INPD / 4) + c];
            v4h hv; hv[0] = (_Float16)v[0]; hv[1] = (_Float16)v[1];
            hv[2] = (_Float16)v[2]; hv[3] = (_Float16)v[3];
            *(v4h*)(x16s + m * INPD + c * 4) = hv;
        }
        __syncthreads();  // new state(t+1) visible

        // ---------------- output projection: y[t] = h_n@outh^T + zr_n@outz^T + b ----
        if (wave < 4) {
            v8f acc = {};
#pragma unroll
            for (int kb = 0; kb < 8; ++kb) {
                const int k0 = kb * 32;
                v16h Boh = bfrag(pOh, k0);
                v16h Boz = bfrag(pOz, k0);
                v16h Ah  = afrag(h16s,  ln, HIDD, k0, grp);
                v16h Azr = afrag(zr16s, ln, HIDD, k0, grp);
                acc = wmma_f16(Ah,  Boh, acc);
                acc = wmma_f16(Azr, Boz, acc);
            }
#pragma unroll
            for (int r = 0; r < 8; ++r) {
                const int m = r + 8 * grp;
                yg[((size_t)t * BATCH + b0 + m) * OUTD + no] = acc[r] + bOut;
            }
        }
        // next iteration's GEMM phase only *reads* LDS; the overwrite happens after
        // the post-GEMM barrier, so out-projection reads are safe without extra sync.
    }
}

extern "C" void kernel_launch(void* const* d_in, const int* in_sizes, int n_in,
                              void* d_out, int out_size, void* d_ws, size_t ws_size,
                              hipStream_t stream) {
    const float* x       = (const float*)d_in[0];
    const float* Wh_w    = (const float*)d_in[1];
    const float* Wh_b    = (const float*)d_in[2];
    const float* Uh_w    = (const float*)d_in[3];
    const float* Uh_b    = (const float*)d_in[4];
    const float* Wtauh_w = (const float*)d_in[5];
    const float* Wtauh_b = (const float*)d_in[6];
    const float* b_h     = (const float*)d_in[7];
    const float* Wzr_w   = (const float*)d_in[8];
    const float* Wzr_b   = (const float*)d_in[9];
    const float* Wzi_w   = (const float*)d_in[10];
    const float* Wzi_b   = (const float*)d_in[11];
    const float* Uz_w    = (const float*)d_in[12];
    const float* Uz_b    = (const float*)d_in[13];
    const float* Wtauz_w = (const float*)d_in[14];
    const float* Wtauz_b = (const float*)d_in[15];
    const float* bzr     = (const float*)d_in[16];
    const float* bzi     = (const float*)d_in[17];
    const float* h2z_w   = (const float*)d_in[18];
    const float* h2z_b   = (const float*)d_in[19];
    const float* z2h_w   = (const float*)d_in[20];
    const float* z2h_b   = (const float*)d_in[21];
    const float* outh_w  = (const float*)d_in[22];
    const float* outh_b  = (const float*)d_in[23];
    const float* outz_w  = (const float*)d_in[24];
    const float* outz_b  = (const float*)d_in[25];
    (void)in_sizes; (void)n_in; (void)out_size; (void)ws_size;

    _Float16* ws16 = (_Float16*)d_ws;
    float*    wsf  = (float*)((char*)d_ws + WS_F16_BYTES);

    ctl_prep_kernel<<<256, 256, 0, stream>>>(Wh_w, Wtauh_w, Wzr_w, Wzi_w, Wtauz_w,
                                             h2z_w, z2h_w, Uh_w, Uz_w, outh_w, outz_w,
                                             Uh_b, b_h, z2h_b, Uz_b, h2z_b, ws16, wsf);

    ctl_recurrent_kernel<<<BATCH / 16, 512, 0, stream>>>(x, ws16, wsf,
                                                         Wh_b, Wtauh_b, Wzr_b, Wzi_b,
                                                         Wtauz_b, bzr, bzi,
                                                         outh_b, outz_b, (float*)d_out);
}
